// SparseGraphAttention_30863634989574
// MI455X (gfx1250) — matrix operand order repaired and checked by reading, not compiled, not run on hardware
//
#include <hip/hip_runtime.h>
#include <hip/hip_bf16.h>

typedef __attribute__((ext_vector_type(16))) _Float16 v16h;
typedef __attribute__((ext_vector_type(8)))  _Float16 v8h;
typedef __attribute__((ext_vector_type(8)))  float    v8f;
typedef __attribute__((ext_vector_type(4)))  float    v4f;

#define LDSK 136   // padded row stride in halves (128 + 8): spreads LDS banks

// ---------------------------------------------------------------------------
// Kernel 1: h = x @ W via v_wmma_f32_16x16x32_f16.
// Block = 256 threads (8 waves). Block covers 64 rows of x (4 M-tiles of 16).
// Wave w -> m_tile = w&3, n-tiles = (w>>2)*4 .. +3.  K=128 -> 4 wmma k-steps.
// W is staged TRANSPOSED in LDS (sW[n][k]) so B fragments are contiguous.
// ---------------------------------------------------------------------------
__global__ __launch_bounds__(256)
void gat_gemm_wmma(const float* __restrict__ x, const float* __restrict__ W,
                   float* __restrict__ h, int R /* = B*N rows */)
{
    __shared__ _Float16 sW[128 * LDSK];  // sW[n*LDSK + k] = W[k, n]  (f16)
    __shared__ _Float16 sA[64  * LDSK];  // sA[r*LDSK + k] = x[row0+r, k] (f16)

    const int tid  = threadIdx.x;
    const int row0 = blockIdx.x * 64;

    // Stage W transposed (coalesced global read, strided LDS write)
    for (int idx = tid; idx < 128 * 128; idx += 256) {
        int k = idx >> 7, n = idx & 127;
        sW[n * LDSK + k] = (_Float16)W[idx];
    }
    // Stage x tile (rows beyond R padded with zeros)
    for (int idx = tid; idx < 64 * 128; idx += 256) {
        int r = idx >> 7, k = idx & 127;
        int grow = row0 + r;
        float v = (grow < R) ? x[(long long)grow * 128 + k] : 0.0f;
        sA[r * LDSK + k] = (_Float16)v;
    }
    __syncthreads();

    const int wave  = tid >> 5;
    const int lane  = tid & 31;
    const int m     = lane & 15;   // row/col within tile
    const int s     = lane >> 4;   // half-wave select
    const int mt    = wave & 3;    // M tile (16 rows)
    const int nbase = (wave >> 2) * 4; // first of 4 N tiles

    v8f acc[4] = {};  // zero accumulators

    const int arow = mt * 16 + m;

    for (int kt = 0; kt < 4; ++kt) {
        const int kb = kt * 32;
        // A fragment, 16x32 f16. CDNA5 layout: lane(m,s) halves j<8 -> K=kb+s*8+j,
        // halves j>=8 -> K=kb+16+s*8+(j-8). Two contiguous 16B LDS loads.
        union { v16h v; v8h p[2]; } af;
        af.p[0] = *(const v8h*)&sA[arow * LDSK + kb + s * 8];
        af.p[1] = *(const v8h*)&sA[arow * LDSK + kb + 16 + s * 8];

        for (int t = 0; t < 4; ++t) {
            const int nt = nbase + t;
            // B fragment, 32x16 f16: lane(n,s) halves j<8 -> K = kb + s*8 + j, j>=8 -> K = kb + 16 + s*8 + (j-8).
            // sW is [n][k] so this is two runs of 8 contiguous halves = two 16B loads.
            const int bn = nt * 16 + m;
            union { v16h v; v8h p[2]; } bf;
            bf.p[0] = *(const v8h*)&sW[bn * LDSK + kb + s * 8];
            bf.p[1] = *(const v8h*)&sW[bn * LDSK + kb + 16 + s * 8];

            acc[t] = __builtin_amdgcn_wmma_f32_16x16x32_f16(
                false, af.v, false, bf.v, (short)0, acc[t], false, false);
        }
    }

    // C/D layout: VGPR r -> lanes0-15: (M=r, N=lane); lanes16-31: (M=r+8, N=lane-16)
    for (int t = 0; t < 4; ++t) {
        const int col = (nbase + t) * 16 + m;
        for (int r = 0; r < 8; ++r) {
            const int mrow = (s == 0) ? r : (r + 8);
            const int grow = row0 + mt * 16 + mrow;
            if (grow < R) h[(long long)grow * 128 + col] = acc[t][r];
        }
    }
}

// ---------------------------------------------------------------------------
// Kernel 2: per-node attention scalars.
// alpha_src[n,h] = dot(h[n,h,:], attn[h,0:32]); alpha_dst with attn[h,32:64].
// One wave per node row: lane covers 4 channels, 8-lane tree-reduce per head.
// ---------------------------------------------------------------------------
__global__ __launch_bounds__(256)
void gat_alpha(const float* __restrict__ h, const float* __restrict__ attn,
               float* __restrict__ a_src, float* __restrict__ a_dst, int R)
{
    const int w    = (blockIdx.x * blockDim.x + threadIdx.x) >> 5;
    const int lane = threadIdx.x & 31;
    if (w >= R) return;

    const int c0   = lane * 4;     // channels [c0, c0+4), all within one head
    const int head = lane >> 3;    // 8 lanes per head (HD=32)

    v4f hv = *(const v4f*)&h[(long long)w * 128 + c0];
    float ps = 0.0f, pd = 0.0f;
    #pragma unroll
    for (int i = 0; i < 4; ++i) {
        const int d = (c0 + i) & 31;
        ps += hv[i] * attn[head * 64 + d];
        pd += hv[i] * attn[head * 64 + 32 + d];
    }
    #pragma unroll
    for (int off = 1; off < 8; off <<= 1) {
        ps += __shfl_xor(ps, off, 32);
        pd += __shfl_xor(pd, off, 32);
    }
    if ((lane & 7) == 0) {
        a_src[w * 4 + head] = ps;
        a_dst[w * 4 + head] = pd;
    }
}

__device__ __forceinline__ float leaky02(float v) {
    return v > 0.0f ? v : 0.2f * v;
}

// float atomic-max via int ordering trick (valid for non-NaN IEEE floats)
__device__ __forceinline__ void atomicMaxF(float* addr, float val) {
    if (val >= 0.0f)
        atomicMax((int*)addr, __float_as_int(val));
    else
        atomicMin((unsigned int*)addr, __float_as_uint(val));
}

// ---------------------------------------------------------------------------
// Kernel 3: scatter-max of leaky-relu scores into attn_max (init 0, like ref).
// One thread per (b, e, head).
// ---------------------------------------------------------------------------
__global__ __launch_bounds__(256)
void gat_edge_max(const int* __restrict__ ei, const float* __restrict__ a_src,
                  const float* __restrict__ a_dst, float* __restrict__ amax,
                  int E, int NH, long long total)
{
    const long long idx = (long long)blockIdx.x * blockDim.x + threadIdx.x;
    if (idx >= total) return;
    const int hh = (int)(idx & 3);
    const long long be = idx >> 2;
    const int b = (int)(be / E);
    const int e = (int)(be - (long long)b * E);
    const int src = ei[e], dst = ei[E + e];
    const float sc = leaky02(a_src[b * NH + src * 4 + hh] +
                             a_dst[b * NH + dst * 4 + hh]);
    atomicMaxF(&amax[b * NH + dst * 4 + hh], sc);
}

// ---------------------------------------------------------------------------
// Kernel 4: scatter-add of exp(score - max[dst]) into attn_sum.
// ---------------------------------------------------------------------------
__global__ __launch_bounds__(256)
void gat_edge_sum(const int* __restrict__ ei, const float* __restrict__ a_src,
                  const float* __restrict__ a_dst, const float* __restrict__ amax,
                  float* __restrict__ asum, int E, int NH, long long total)
{
    const long long idx = (long long)blockIdx.x * blockDim.x + threadIdx.x;
    if (idx >= total) return;
    const int hh = (int)(idx & 3);
    const long long be = idx >> 2;
    const int b = (int)(be / E);
    const int e = (int)(be - (long long)b * E);
    const int src = ei[e], dst = ei[E + e];
    const float sc = leaky02(a_src[b * NH + src * 4 + hh] +
                             a_dst[b * NH + dst * 4 + hh]);
    const float m = amax[b * NH + dst * 4 + hh];
    atomicAdd(&asum[b * NH + dst * 4 + hh], __expf(sc - m));
}

// ---------------------------------------------------------------------------
// Kernel 5: message aggregation.
// One wave per (b, e): lane covers 4 channels (float4 gather of h_src from L2),
// recomputes its head's normalized coefficient, 4 float atomic adds to out[dst].
// ---------------------------------------------------------------------------
__global__ __launch_bounds__(256)
void gat_edge_aggr(const int* __restrict__ ei, const float* __restrict__ h,
                   const float* __restrict__ a_src, const float* __restrict__ a_dst,
                   const float* __restrict__ amax, const float* __restrict__ asum,
                   float* __restrict__ out, int E, int N, long long totalWaves)
{
    const long long w = ((long long)blockIdx.x * blockDim.x + threadIdx.x) >> 5;
    if (w >= totalWaves) return;
    const int lane = threadIdx.x & 31;
    const int b = (int)(w / E);
    const int e = (int)(w - (long long)b * E);
    const int src = ei[e], dst = ei[E + e];
    const int head = lane >> 3;
    const int NH = N * 4;

    const float sc = leaky02(a_src[b * NH + src * 4 + head] +
                             a_dst[b * NH + dst * 4 + head]);
    const float m  = amax[b * NH + dst * 4 + head];
    const float ss = asum[b * NH + dst * 4 + head];
    const float coeff = __expf(sc - m) / (ss + 1e-8f);

    const int c0 = lane * 4;
    const long long hbase = ((long long)b * N + src) * 128 + c0;
    const long long obase = ((long long)b * N + dst) * 128 + c0;
    v4f hv = *(const v4f*)&h[hbase];
    atomicAdd(&out[obase + 0], hv[0] * coeff);
    atomicAdd(&out[obase + 1], hv[1] * coeff);
    atomicAdd(&out[obase + 2], hv[2] * coeff);
    atomicAdd(&out[obase + 3], hv[3] * coeff);
}

// ---------------------------------------------------------------------------
extern "C" void kernel_launch(void* const* d_in, const int* in_sizes, int n_in,
                              void* d_out, int out_size, void* d_ws, size_t ws_size,
                              hipStream_t stream) {
    const float* x    = (const float*)d_in[0];
    const int*   ei   = (const int*)d_in[1];   // int32 (jax x64 disabled)
    const float* W    = (const float*)d_in[2];
    const float* attn = (const float*)d_in[3];
    float*       out  = (float*)d_out;

    const int R = in_sizes[0] / 128;   // B*N rows
    const int E = in_sizes[1] / 2;
    const int N = 50000;
    const int B = R / N;
    const int NH = N * 4;

    // Scratch layout in d_ws (~54.4 MB):
    char* ws = (char*)d_ws;
    float* h     = (float*)ws;                                   // R*128 f32
    float* a_src = (float*)(ws + (size_t)R * 128 * sizeof(float));
    float* a_dst = a_src + (size_t)R * 4;
    float* amax  = a_dst + (size_t)R * 4;
    float* asum  = amax  + (size_t)R * 4;

    // Zero out / softmax state each call (capture-legal async memsets).
    hipMemsetAsync(out,  0, (size_t)out_size * sizeof(float), stream);
    hipMemsetAsync(amax, 0, (size_t)R * 4 * sizeof(float), stream);
    hipMemsetAsync(asum, 0, (size_t)R * 4 * sizeof(float), stream);

    // 1) GEMM: 64 rows per block
    {
        dim3 grid((R + 63) / 64);
        gat_gemm_wmma<<<grid, 256, 0, stream>>>(x, W, h, R);
    }
    // 2) per-node alpha scalars: one wave per row
    {
        long long threads = (long long)R * 32;
        dim3 grid((unsigned)((threads + 255) / 256));
        gat_alpha<<<grid, 256, 0, stream>>>(h, attn, a_src, a_dst, R);
    }
    // 3) scatter-max
    {
        long long total = (long long)B * E * 4;
        dim3 grid((unsigned)((total + 255) / 256));
        gat_edge_max<<<grid, 256, 0, stream>>>(ei, a_src, a_dst, amax, E, NH, total);
    }
    // 4) scatter-sum of exp
    {
        long long total = (long long)B * E * 4;
        dim3 grid((unsigned)((total + 255) / 256));
        gat_edge_sum<<<grid, 256, 0, stream>>>(ei, a_src, a_dst, amax, asum, E, NH, total);
    }
    // 5) aggregation: one wave per (b, e)
    {
        long long waves = (long long)B * E;
        dim3 grid((unsigned)((waves * 32 + 255) / 256));
        gat_edge_aggr<<<grid, 256, 0, stream>>>(ei, h, a_src, a_dst, amax, asum,
                                                out, E, N, waves);
    }
}
